// GCN_5342939316765
// MI455X (gfx1250) — matrix operand order repaired and checked by reading, not compiled, not run on hardware
//
#include <hip/hip_runtime.h>
#include <hip/hip_bf16.h>

// ---------- types for WMMA ----------
typedef __bf16 bf16_t;
typedef bf16_t v16bf __attribute__((ext_vector_type(16)));
typedef float  v8f   __attribute__((ext_vector_type(8)));
typedef unsigned int v8u __attribute__((ext_vector_type(8)));

// GCC-style vector type matching the async-to-LDS builtin signature
typedef int gv4i __attribute__((vector_size(16)));
typedef gv4i __attribute__((address_space(1)))* gptr_v4i;
typedef gv4i __attribute__((address_space(3)))* lptr_v4i;

#if __has_builtin(__builtin_amdgcn_global_load_async_to_lds_b128)
#define HAS_ASYNC_LDS 1
#else
#define HAS_ASYNC_LDS 0
#endif

__device__ __forceinline__ void wait_async0() {
#if __has_builtin(__builtin_amdgcn_s_wait_asynccnt)
  __builtin_amdgcn_s_wait_asynccnt(0);
#else
  asm volatile("s_wait_asynccnt 0x0" ::: "memory");
#endif
}

__device__ __forceinline__ unsigned int f2bf(float f) {
  unsigned int u = __builtin_bit_cast(unsigned int, f);
  u += 0x7FFFu + ((u >> 16) & 1u);   // round-to-nearest-even
  return u >> 16;
}
__device__ __forceinline__ unsigned int pack2bf(float lo, float hi) {
  return f2bf(lo) | (f2bf(hi) << 16);
}

// ---------- small utility kernels ----------
__global__ void fill_f32(float* __restrict__ p, long long n, float v) {
  long long i = (long long)blockIdx.x * blockDim.x + threadIdx.x;
  if (i < n) p[i] = v;
}

__global__ void deg_accum(const int* __restrict__ dst, float* __restrict__ deg, int E) {
  int i = blockIdx.x * blockDim.x + threadIdx.x;
  if (i < E) atomicAdd(&deg[dst[i]], 1.0f);
}

__global__ void rsqrt_k(const float* __restrict__ deg, float* __restrict__ dis, int n) {
  int i = blockIdx.x * blockDim.x + threadIdx.x;
  if (i < n) dis[i] = __frsqrt_rn(deg[i]);   // deg >= 1 (self-loop), no zero case
}

// ---------- pack W [K x COLS] f32 into per-lane WMMA B-fragment order ----------
// Fragment order (contiguous k, same as gemm_bf16's A reads; bf16 B 32x16): lane<16 -> K base 0, lane>=16 -> K base 16;
// VGPR v holds (K=kb+2v, K=kb+2v+1) for column ct*16 + (lane&15).
__global__ void pack_bfrag(const float* __restrict__ W, unsigned int* __restrict__ out,
                           int K, int COLS) {
  int i = blockIdx.x * blockDim.x + threadIdx.x;
  int total = (K / 32) * (COLS / 16) * 32 * 8;
  if (i >= total) return;
  int v    = i & 7;
  int lane = (i >> 3) & 31;
  int ct   = (i >> 8) % (COLS / 16);
  int ks   = (i >> 8) / (COLS / 16);
  int col  = ct * 16 + (lane & 15);
  int k    = ks * 32 + ((lane < 16) ? 0 : 16) + 2 * v;
  out[i] = pack2bf(W[k * COLS + col], W[(k + 1) * COLS + col]);
}

// ---------- WMMA GEMM: hs = agg = dis[row] * (A @ W) ----------
// One block = 16-row strip; COLS/16 waves, each owns one 16x16 tile across all K.
// A: [M, K] (f32 if !ABF, bf16-as-u16 if ABF). Bfrag: pre-packed fragments (L2-resident).
// K-tile = 64 (2 WMMA k-steps per stage), double-buffered LDS for A.
// Epilogue writes the scaled result to BOTH hs (gather source) and agg (accumulator
// pre-seeded with the self-loop term), eliminating the zero-fill pass.
template <int COLS, bool ABF, int K>
__global__ __launch_bounds__(2 * COLS)
void gemm_bf16(const void* __restrict__ Av, const unsigned int* __restrict__ Bfrag,
               const float* __restrict__ dis, float* __restrict__ hs,
               float* __restrict__ agg) {
  constexpr int NT    = 2 * COLS;    // threads = 32 * (COLS/16)
  constexpr int CT    = COLS / 16;   // column tiles
  constexpr int NSTEP = K / 64;      // 64-wide K chunks
  __shared__ unsigned short sA[2][16][72];   // 16 x 64 bf16, pitch 72 (rows 16B-aligned)

  const int tid  = threadIdx.x;
  const int lane = tid & 31;
  const int wv   = tid >> 5;
  const int row0 = blockIdx.x * 16;

  // Fragment geometry (wave32):
  const int arow = lane & 15;                // A row within tile
  const int akb  = (lane < 16) ? 0 : 16;     // A K-base for lane half
  const int bcol = (wv << 4) + (lane & 15);  // B/C column

  // ---- A staging ----
  auto stageA = [&](int buf, int kt0) {
    if (!ABF) {
      // f32 source: vector load + in-register bf16 convert + 8B LDS store
      const float* A = (const float*)Av;
      for (int i = tid; i < 256; i += NT) {          // 16 rows x 16 float4
        int r = i >> 4, c4 = (i & 15) << 2;
        const float4 g = *(const float4*)&A[(long long)(row0 + r) * K + kt0 + c4];
        uint2 u;
        u.x = pack2bf(g.x, g.y);
        u.y = pack2bf(g.z, g.w);
        *(uint2*)&sA[buf][r][c4] = u;
      }
    } else {
      // bf16 source: raw copy -> gfx1250 async global->LDS (b128 per thread)
      const unsigned short* A = (const unsigned short*)Av;
      for (int i = tid; i < 128; i += NT) {          // 16 rows x 8 chunks of 8 bf16 (16B)
        int r = i >> 3, c8 = (i & 7) << 3;
        const unsigned short* g = &A[(long long)(row0 + r) * K + kt0 + c8];
#if HAS_ASYNC_LDS
        __builtin_amdgcn_global_load_async_to_lds_b128(
            (gptr_v4i)(unsigned long long)g,
            (lptr_v4i)(unsigned int)(unsigned long long)&sA[buf][r][c8],
            0, 0);
#else
        const uint2 gv = *(const uint2*)g;
        *(uint2*)&sA[buf][r][c8] = gv;
        const uint2 gv2 = *(const uint2*)(g + 4);
        *(uint2*)&sA[buf][r][c8 + 4] = gv2;
#endif
      }
    }
  };

  v8f acc = {};
  stageA(0, 0);

  for (int t = 0; t < NSTEP; ++t) {
    if (ABF && HAS_ASYNC_LDS) wait_async0();      // async LDS writes visible
    __syncthreads();                              // buf (t&1) ready for reads
    if (t + 1 < NSTEP) stageA((t + 1) & 1, (t + 1) * 64);
    if (!ABF && t + 2 < NSTEP) {                  // stream-prefetch two tiles ahead
      const float* A = (const float*)Av;
      __builtin_prefetch(&A[(long long)(row0 + (tid >> 4)) * K + (t + 2) * 64 + ((tid & 15) << 2)], 0, 3);
    }
#pragma unroll
    for (int s = 0; s < 2; ++s) {                 // two 32-wide k-steps per tile
      const int buf = t & 1, koff = s << 5;
      // A fragment: 8 dword LDS reads (pairs of bf16 K-contiguous)
      v8u au;
      const unsigned int* pa = (const unsigned int*)&sA[buf][arow][koff];
#pragma unroll
      for (int v = 0; v < 8; ++v) {
        int k = akb + 2 * v;
        au[v] = pa[k >> 1];
      }
      // B fragment: one aligned 32B lane-contiguous load from packed weights
      const int ks = t * 2 + s;
      const v8u bu = *(const v8u*)&Bfrag[(((ks * CT + wv) << 5) + lane) << 3];

      v16bf a = __builtin_bit_cast(v16bf, au);
      v16bf b = __builtin_bit_cast(v16bf, bu);
      acc = __builtin_amdgcn_wmma_f32_16x16x32_bf16(
          false, a, false, b, (short)0, acc, false, false);
    }
  }

  // ---- epilogue: row-scale by dis; seed agg with the self-loop term ----
#pragma unroll
  for (int r = 0; r < 8; ++r) {
    int rl   = r + ((lane < 16) ? 0 : 8);   // C layout: VGPRr = M=r / M=r+8
    int grow = row0 + rl;
    long long idx = (long long)grow * COLS + bcol;
    float v = dis[grow] * acc[r];
    hs[idx]  = v;
    agg[idx] = v;
  }
}

// ---------- edge scatter-add: agg[dst] += hs[src]  (C = power of two) ----------
__global__ void scatter_add(const float* __restrict__ hs, const int* __restrict__ src,
                            const int* __restrict__ dst, float* __restrict__ agg,
                            int E, int C, int chunkShift) {
  int t = blockIdx.x * blockDim.x + threadIdx.x;   // one thread = (edge, 4 channels)
  int e = t >> chunkShift;
  if (e >= E) return;
  int c4 = (t & ((1 << chunkShift) - 1)) << 2;
  int s = src[e], d = dst[e];
  const float4 v = *(const float4*)&hs[(long long)s * C + c4];
  float* p = &agg[(long long)d * C + c4];
  atomicAdd(p + 0, v.x);
  atomicAdd(p + 1, v.y);
  atomicAdd(p + 2, v.z);
  atomicAdd(p + 3, v.w);
}

// ---------- finalize layer 1: h = bf16(relu(dis*agg + b)) ----------
__global__ void finalize1(const float* __restrict__ agg, const float* __restrict__ dis,
                          const float* __restrict__ b, unsigned short* __restrict__ hout,
                          long long n, int cmask, int cshift) {
  long long i = (long long)blockIdx.x * blockDim.x + threadIdx.x;
  if (i >= n) return;
  int row = (int)(i >> cshift);
  int c   = (int)i & cmask;
  float v = dis[row] * agg[i] + b[c];
  v = fmaxf(v, 0.0f);
  hout[i] = (unsigned short)f2bf(v);
}

// ---------- finalize layer 2: out = dis*agg + b ----------
__global__ void finalize2(const float* __restrict__ agg, const float* __restrict__ dis,
                          const float* __restrict__ b, float* __restrict__ out,
                          long long n, int cmask, int cshift) {
  long long i = (long long)blockIdx.x * blockDim.x + threadIdx.x;
  if (i >= n) return;
  int row = (int)(i >> cshift);
  int c   = (int)i & cmask;
  out[i] = dis[row] * agg[i] + b[c];
}

// ---------- launcher ----------
extern "C" void kernel_launch(void* const* d_in, const int* in_sizes, int n_in,
                              void* d_out, int out_size, void* d_ws, size_t ws_size,
                              hipStream_t stream) {
  const float* x  = (const float*)d_in[0];
  const int*   ei = (const int*)d_in[1];     // edge_index, flat [2, E]
  const float* W1 = (const float*)d_in[2];
  const float* b1 = (const float*)d_in[3];
  const float* W2 = (const float*)d_in[4];
  const float* b2 = (const float*)d_in[5];

  const int Ch   = in_sizes[3];              // 128
  const int Cout = in_sizes[5];              // 64
  const int Cin  = in_sizes[2] / Ch;         // 512
  const int N    = in_sizes[0] / Cin;        // 50000
  const int E    = in_sizes[1] / 2;          // 800000
  const int* srcI = ei;
  const int* dstI = ei + E;

  // workspace carve-out (256B aligned)
  char* ws = (char*)d_ws;
  size_t o = 0;
  auto carve = [&](size_t bytes) {
    char* p = ws + o;
    o = (o + bytes + 255) & ~(size_t)255;
    return p;
  };
  float*        deg  = (float*)carve((size_t)N * 4);
  float*        dis  = (float*)carve((size_t)N * 4);
  unsigned int* w1f  = (unsigned int*)carve((size_t)(Cin / 32) * (Ch / 16) * 256 * 4);
  unsigned int* w2f  = (unsigned int*)carve((size_t)(Ch / 32) * (Cout / 16) * 256 * 4);
  float*        hs1  = (float*)carve((size_t)N * Ch * 4);
  float*        agg1 = (float*)carve((size_t)N * Ch * 4);
  unsigned short* h2b = (unsigned short*)carve((size_t)N * Ch * 2);
  float*        hs2  = (float*)carve((size_t)N * Cout * 4);
  float*        agg2 = (float*)carve((size_t)N * Cout * 4);
  (void)ws_size; (void)n_in; (void)out_size;

  auto cdiv = [](long long a, long long b) { return (int)((a + b - 1) / b); };
  auto ilog2 = [](int v) { int s = 0; while ((1 << s) < v) ++s; return s; };

  const long long n1 = (long long)N * Ch;
  const long long n2 = (long long)N * Cout;

  // normalization (deg starts at 1.0 = self-loop)
  fill_f32<<<cdiv(N, 256), 256, 0, stream>>>(deg, N, 1.0f);
  deg_accum<<<cdiv(E, 256), 256, 0, stream>>>(dstI, deg, E);
  rsqrt_k<<<cdiv(N, 256), 256, 0, stream>>>(deg, dis, N);

  // weights -> per-lane WMMA fragment order (L2-resident)
  pack_bfrag<<<cdiv((Cin / 32) * (Ch / 16) * 256, 256), 256, 0, stream>>>(W1, w1f, Cin, Ch);
  pack_bfrag<<<cdiv((Ch / 32) * (Cout / 16) * 256, 256), 256, 0, stream>>>(W2, w2f, Ch, Cout);

  // layer 1: hs1 = agg1 = dis[row] * (x @ W1); agg1 pre-seeded with self-loop term
  gemm_bf16<128, false, 512><<<N / 16, 256, 0, stream>>>(x, w1f, dis, hs1, agg1);
  scatter_add<<<cdiv((long long)E * (Ch / 4), 256), 256, 0, stream>>>(
      hs1, srcI, dstI, agg1, E, Ch, ilog2(Ch / 4));
  finalize1<<<cdiv(n1, 256), 256, 0, stream>>>(agg1, dis, b1, h2b, n1, Ch - 1, ilog2(Ch));

  // layer 2: hs2 = agg2 = dis[row] * (h @ W2)
  gemm_bf16<64, true, 128><<<N / 16, 128, 0, stream>>>(h2b, w2f, dis, hs2, agg2);
  scatter_add<<<cdiv((long long)E * (Cout / 4), 256), 256, 0, stream>>>(
      hs2, srcI, dstI, agg2, E, Cout, ilog2(Cout / 4));
  finalize2<<<cdiv(n2, 256), 256, 0, stream>>>(agg2, dis, b2, (float*)d_out, n2,
                                               Cout - 1, ilog2(Cout));
}
